// CKGGCN_65025804861451
// MI455X (gfx1250) — matrix operand order repaired and checked by reading, not compiled, not run on hardware
//
#include <hip/hip_runtime.h>
#include <hip/hip_bf16.h>

typedef _Float16 h16;
typedef __attribute__((ext_vector_type(16))) _Float16 v16h;
typedef __attribute__((ext_vector_type(8)))  float    v8f;

#define N_USERS   50000
#define N_ENTITIES 100000
#define DIMS      64
#define E_KG      1500000L
#define E_IN      1000000L
#define RSQRT_DK  0.17677669529663687f   // 1/sqrt(32)

// ---- monotone float<->uint encoding for atomic max on signed floats ----
__device__ __forceinline__ unsigned ford_enc(float f) {
  unsigned u = __float_as_uint(f);
  return (u >> 31) ? ~u : (u | 0x80000000u);
}
__device__ __forceinline__ float ford_dec(unsigned e) {
  return (e >> 31) ? __uint_as_float(e & 0x7FFFFFFFu) : __uint_as_float(~e);
}

// ======================= prep: Wt[n][k] = (f16) W_Q[k][n] =======================
__global__ void k_prepw(const float* __restrict__ W, h16* __restrict__ Wt) {
  int i = blockIdx.x * 256 + threadIdx.x;
  if (i < 64 * 64) {
    int n = i >> 6, k = i & 63;
    Wt[i] = (h16)W[k * 64 + n];
  }
}

// ======================= per-layer zero =======================
__global__ void k_zero(float* __restrict__ ent, float* __restrict__ usr,
                       float* __restrict__ segsum, unsigned* __restrict__ segmax) {
  long gid = (long)blockIdx.x * blockDim.x + threadIdx.x;
  if (gid < (long)N_ENTITIES * DIMS) ent[gid] = 0.f;
  if (gid < (long)N_USERS * DIMS)    usr[gid] = 0.f;
  if (gid < (long)N_ENTITIES * 2) { segsum[gid] = 0.f; segmax[gid] = 0u; }
}

// ======================= WMMA attention-score kernel =======================
// 128 threads = 4 waves; each wave owns a 16-edge tile.
// Q = head_emb @ W, Kr = tail_emb @ W via v_wmma_f32_16x16x32_f16,
// score[e][h] = (1/sqrt(dk)) * sum_d Q[e][h*32+d]*Kr[e][h*32+d]*rel[e][h*32+d]
__global__ __launch_bounds__(128) void k_attn(
    const float* __restrict__ ent, const int* __restrict__ eidx,
    const int* __restrict__ etype, const float* __restrict__ rel_emb,
    const h16* __restrict__ Wt, float* __restrict__ score,
    unsigned* __restrict__ segmax) {
  __shared__ h16  lA[4][16][72];   // head rows, f16, padded (bank-conflict break)
  __shared__ h16  lB[4][16][72];   // tail rows
  __shared__ h16  lW[64][72];      // Wt (n-major), padded
  __shared__ float lQ[4][16][68];  // Q accumulators spilled for score pass
  __shared__ float lK[4][16][68];  // Kraw accumulators

  const int tid  = threadIdx.x;
  const int wave = tid >> 5;
  const int lane = tid & 31;
  const long e0  = (long)blockIdx.x * 64 + (long)wave * 16;

  // stage Wt into LDS (once per block)
  for (int i = tid; i < 64 * 64; i += 128) {
    int n = i >> 6, k = i & 63;
    lW[n][k] = Wt[i];
  }

  // gather + f32->f16: lane covers row r = lane/2, cols c0 = (lane&1)*32
  {
    int r  = lane >> 1;
    int c0 = (lane & 1) * 32;
    long e = e0 + r;
    if (e < E_KG) {
      long hi = eidx[e];
      long ti = eidx[E_KG + e];
      const float4* hp = (const float4*)(ent + hi * DIMS + c0);
      const float4* tp = (const float4*)(ent + ti * DIMS + c0);
      for (int j = 0; j < 8; ++j) {
        float4 a = hp[j], b = tp[j];
        int c = c0 + 4 * j;
        lA[wave][r][c+0]=(h16)a.x; lA[wave][r][c+1]=(h16)a.y;
        lA[wave][r][c+2]=(h16)a.z; lA[wave][r][c+3]=(h16)a.w;
        lB[wave][r][c+0]=(h16)b.x; lB[wave][r][c+1]=(h16)b.y;
        lB[wave][r][c+2]=(h16)b.z; lB[wave][r][c+3]=(h16)b.w;
      }
    } else {
      for (int j = 0; j < 32; ++j) {
        lA[wave][r][c0 + j] = (h16)0.f;
        lB[wave][r][c0 + j] = (h16)0.f;
      }
    }
  }
  __syncthreads();

  union V16 { uint4 q[2]; v16h v; };

  // A operand layout (16x32 f16): lane m=L&15; halves 0-7 = K kb..kb+7,
  // halves 8-15 = K 16+kb..16+kb+7, kb = (L<16)?0:8
  const int m  = lane & 15;
  const int kb = (lane < 16) ? 0 : 8;
  v16h aH[2], aT[2];
  for (int ks = 0; ks < 2; ++ks) {
    V16 u;
    u.q[0] = *(const uint4*)&lA[wave][m][ks * 32 + kb];
    u.q[1] = *(const uint4*)&lA[wave][m][ks * 32 + 16 + kb];
    aH[ks] = u.v;
    u.q[0] = *(const uint4*)&lB[wave][m][ks * 32 + kb];
    u.q[1] = *(const uint4*)&lB[wave][m][ks * 32 + 16 + kb];
    aT[ks] = u.v;
  }

  // B operand (32x16 f16): lane n=L&15; halves 0-7 = K kbB..kbB+7, halves 8-15 = K 16+kbB..16+kbB+7
  const int n   = lane & 15;
  const int kbB = (lane < 16) ? 0 : 8;

  for (int nt = 0; nt < 4; ++nt) {
    v8f accQ = {};
    v8f accK = {};
    for (int ks = 0; ks < 2; ++ks) {
      V16 ub;
      ub.q[0] = *(const uint4*)&lW[nt * 16 + n][ks * 32 + kbB];
      ub.q[1] = *(const uint4*)&lW[nt * 16 + n][ks * 32 + 16 + kbB];
      v16h b = ub.v;
      accQ = __builtin_amdgcn_wmma_f32_16x16x32_f16(false, aH[ks], false, b,
                                                    (short)0, accQ, false, false);
      accK = __builtin_amdgcn_wmma_f32_16x16x32_f16(false, aT[ks], false, b,
                                                    (short)0, accK, false, false);
    }
    // C/D layout: reg r -> M = r + 8*(lane>=16), N = (lane&15) + nt*16
    int mrow = (lane >> 4) * 8;
    for (int r = 0; r < 8; ++r) {
      lQ[wave][mrow + r][nt * 16 + n] = accQ[r];
      lK[wave][mrow + r][nt * 16 + n] = accK[r];
    }
  }
  __syncthreads();

  // score pass: lane covers (edge r = lane&15, head h = lane>>4)
  {
    int r = lane & 15;
    int h = lane >> 4;
    long e = e0 + r;
    if (e < E_KG) {
      int rt = etype[e] - 1;
      const float* rp = rel_emb + (long)rt * DIMS + h * 32;
      float s = 0.f;
      for (int d = 0; d < 32; ++d)
        s += lQ[wave][r][h * 32 + d] * lK[wave][r][h * 32 + d] * rp[d];
      s *= RSQRT_DK;
      score[e * 2 + h] = s;
      int head = eidx[e];
      atomicMax(&segmax[(long)head * 2 + h], ford_enc(s));
    }
  }
}

// ======================= exp + segment sum =======================
__global__ void k_expsum(float* __restrict__ score, const int* __restrict__ eidx,
                         const unsigned* __restrict__ segmax,
                         float* __restrict__ segsum) {
  long gid = (long)blockIdx.x * blockDim.x + threadIdx.x;
  if (gid >= 2 * E_KG) return;
  long e = gid >> 1;
  int h = (int)(gid & 1);
  int head = eidx[e];
  float mx = ford_dec(segmax[(long)head * 2 + h]);
  float ex = __expf(score[gid] - mx);
  score[gid] = ex;
  atomicAdd(&segsum[(long)head * 2 + h], ex);
}

// ======================= attention message scatter =======================
__global__ void k_message(const float* __restrict__ ent, const int* __restrict__ eidx,
                          const int* __restrict__ etype, const float* __restrict__ rel_emb,
                          const float* __restrict__ score, const float* __restrict__ segsum,
                          float* __restrict__ entagg) {
  long gid = (long)blockIdx.x * blockDim.x + threadIdx.x;
  if (gid >= E_KG * 8) return;
  long e = gid >> 3;
  int c0 = (int)(gid & 7) * 8;
  int head = eidx[e];
  int tail = eidx[E_KG + e];
  int rt = etype[e] - 1;
  int h = c0 >> 5;
  float attn = score[e * 2 + h] / segsum[(long)head * 2 + h];
  const float* tp = ent + (long)tail * DIMS + c0;
  const float* rp = rel_emb + (long)rt * DIMS + c0;
  float* op = entagg + (long)head * DIMS + c0;
  for (int d = 0; d < 8; ++d)
    atomicAdd(&op[d], tp[d] * rp[d] * attn);
}

// ======================= row-wise L2 normalize (wave per row) =======================
__global__ __launch_bounds__(256) void k_norm(float* __restrict__ ent, int nrows) {
  int row  = blockIdx.x * 8 + (threadIdx.x >> 5);
  int lane = threadIdx.x & 31;
  if (row >= nrows) return;
  float* p = ent + (long)row * DIMS;
  float a = p[lane], b = p[lane + 32];
  float ss = a * a + b * b;
  for (int mseg = 16; mseg; mseg >>= 1) ss += __shfl_xor(ss, mseg, 32);
  float sc = 1.f / fmaxf(sqrtf(ss), 1e-12f);
  p[lane] = a * sc;
  p[lane + 32] = b * sc;
}

// ======================= weighted bipartite scatter =======================
__global__ void k_scatter(const float* __restrict__ src, const int* __restrict__ idx_src,
                          const int* __restrict__ idx_dst, const float* __restrict__ w,
                          float* __restrict__ dst, long nedge) {
  long gid = (long)blockIdx.x * blockDim.x + threadIdx.x;
  if (gid >= nedge * 8) return;
  long e = gid >> 3;
  int c0 = (int)(gid & 7) * 8;
  long s = idx_src[e];
  long d = idx_dst[e];
  float ww = w[e];
  const float* sp = src + s * DIMS + c0;
  float* dp = dst + d * DIMS + c0;
  for (int j = 0; j < 8; ++j)
    atomicAdd(&dp[j], ww * sp[j]);
}

// ======================= output init / accumulate / finalize =======================
__global__ void k_initout(float* __restrict__ out, const float* __restrict__ uemb,
                          const float* __restrict__ eemb) {
  long gid = (long)blockIdx.x * blockDim.x + threadIdx.x;
  const long nu = (long)N_USERS * DIMS;
  const long nt = nu + (long)N_ENTITIES * DIMS;
  if (gid < nu) out[gid] = uemb[gid];
  else if (gid < nt) out[gid] = eemb[gid - nu];
}

__global__ void k_acc(float* __restrict__ out, const float* __restrict__ usr,
                      const float* __restrict__ ent) {
  long gid = (long)blockIdx.x * blockDim.x + threadIdx.x;
  const long nu = (long)N_USERS * DIMS;
  const long nt = nu + (long)N_ENTITIES * DIMS;
  if (gid < nu) out[gid] += usr[gid];
  else if (gid < nt) out[gid] += ent[gid - nu];
}

__global__ void k_scale(float* __restrict__ out) {
  long gid = (long)blockIdx.x * blockDim.x + threadIdx.x;
  const long nt = (long)(N_USERS + N_ENTITIES) * DIMS;
  if (gid < nt) out[gid] *= (1.0f / 3.0f);
}

// ======================= host launcher =======================
extern "C" void kernel_launch(void* const* d_in, const int* in_sizes, int n_in,
                              void* d_out, int out_size, void* d_ws, size_t ws_size,
                              hipStream_t stream) {
  (void)in_sizes; (void)n_in; (void)out_size; (void)ws_size;
  const float* user_emb   = (const float*)d_in[1];
  const float* entity_emb = (const float*)d_in[2];
  const int*   inter_edge = (const int*)d_in[3];    // [2, E_IN] row-major
  const float* inter_w    = (const float*)d_in[4];
  const int*   edge_index = (const int*)d_in[5];    // [2, E_KG] row-major
  const int*   edge_type  = (const int*)d_in[6];
  const float* rel_emb    = (const float*)d_in[7];
  const float* W_Q        = (const float*)d_in[8];
  float* out = (float*)d_out;

  // workspace carve-up (floats)
  float* ws = (float*)d_ws;
  float*    entA   = ws;                      // 6,400,000
  float*    entB   = entA + 6400000L;         // 6,400,000
  float*    usrA   = entB + 6400000L;         // 3,200,000
  float*    usrB   = usrA + 3200000L;         // 3,200,000
  float*    score  = usrB + 3200000L;         // 3,000,000
  float*    segsum = score + 3000000L;        //   200,000
  unsigned* segmax = (unsigned*)(segsum + 200000L); // 200,000
  h16*      Wt     = (h16*)(segsum + 400000L);      // 4096 halves

  const long out_total = (long)(N_USERS + N_ENTITIES) * DIMS; // 9.6M
  const int gOut   = (int)((out_total + 255) / 256);
  const int gZero  = (int)(((long)N_ENTITIES * DIMS + 255) / 256);
  const int gAttn  = (int)((E_KG + 63) / 64);
  const int gExp   = (int)((2 * E_KG + 255) / 256);
  const int gMsg   = (int)((E_KG * 8 + 255) / 256);
  const int gNorm  = (N_ENTITIES + 7) / 8;
  const int gScat  = (int)((E_IN * 8 + 255) / 256);

  k_prepw<<<16, 256, 0, stream>>>(W_Q, Wt);
  k_initout<<<gOut, 256, 0, stream>>>(out, user_emb, entity_emb);

  const float* ent_cur = entity_emb;
  const float* usr_cur = user_emb;
  float* ent_buf[2] = {entA, entB};
  float* usr_buf[2] = {usrA, usrB};

  for (int l = 0; l < 2; ++l) {  // layers_num == 2 (reference constant)
    float* ent_next = ent_buf[l & 1];
    float* usr_next = usr_buf[l & 1];

    k_zero<<<gZero, 256, 0, stream>>>(ent_next, usr_next, segsum, segmax);
    k_attn<<<gAttn, 128, 0, stream>>>(ent_cur, edge_index, edge_type, rel_emb,
                                      Wt, score, segmax);
    k_expsum<<<gExp, 256, 0, stream>>>(score, edge_index, segmax, segsum);
    k_message<<<gMsg, 256, 0, stream>>>(ent_cur, edge_index, edge_type, rel_emb,
                                        score, segsum, ent_next);
    k_norm<<<gNorm, 256, 0, stream>>>(ent_next, N_ENTITIES);
    // user <- item
    k_scatter<<<gScat, 256, 0, stream>>>(ent_cur, inter_edge + E_IN, inter_edge,
                                         inter_w, usr_next, E_IN);
    // entity <- user
    k_scatter<<<gScat, 256, 0, stream>>>(usr_cur, inter_edge, inter_edge + E_IN,
                                         inter_w, ent_next, E_IN);
    k_acc<<<gOut, 256, 0, stream>>>(out, usr_next, ent_next);

    ent_cur = ent_next;
    usr_cur = usr_next;
  }
  k_scale<<<gOut, 256, 0, stream>>>(out);
}
